// nearAttention_84782654423390
// MI455X (gfx1250) — compile-verified, matrix operand order checked by reading, not run on hardware
//
#include <hip/hip_runtime.h>

// nearAttention for MI455X (gfx1250), wave32 + WMMA f32_16x16x32_f16.
// Shapes hardcoded to the reference: B=4, H=6, L=1024, d=64.

#define B_ 4
#define H_ 6
#define L_ 1024
#define D_ 64
#define BH_ (B_ * H_)

typedef __attribute__((ext_vector_type(16))) _Float16 v16h;
typedef __attribute__((ext_vector_type(8)))  float    v8f;

// ---------------- workspace layout (floats) ----------------
#define WS_K2       0                  // BH_*L_  : ||K_i||^2
#define WS_Q2       (BH_ * L_)         // BH_*L_  : ||Q_j||^2
#define WS_ACC      (2 * BH_ * L_)     // accumulators / stats below
#define WS_BN1_SUM  (WS_ACC + 0 * H_)
#define WS_BN1_SQ   (WS_ACC + 1 * H_)
#define WS_BN1_MEAN (WS_ACC + 2 * H_)
#define WS_BN1_RSTD (WS_ACC + 3 * H_)
#define WS_BNK_SUM  (WS_ACC + 4 * H_)
#define WS_BNK_SQ   (WS_ACC + 5 * H_)
#define WS_BNQ_SUM  (WS_ACC + 6 * H_)
#define WS_BNQ_SQ   (WS_ACC + 7 * H_)
#define WS_ACC_N    (8 * H_)

#define WMMA_F16(A, Bm, C) \
  __builtin_amdgcn_wmma_f32_16x16x32_f16(false, (A), false, (Bm), (short)0, (C), false, false)

// ---- fragment loaders (ISA 7.12.2 layouts, wave32) ----
// A (16x32 MxK, row-major source): lane<16 -> row=lane, K 0..7 & 16..23;
// lane>=16 -> row=lane-16, K 8..15 & 24..31.  Also used for B when the
// row-major source holds B^T (columns of B are contiguous rows).
__device__ __forceinline__ v16h frag_row(const _Float16* base, int ld, int lane) {
  int m  = lane & 15;
  int ko = (lane & 16) ? 8 : 0;
  const _Float16* p = base + m * ld + ko;
  v16h r;
#pragma unroll
  for (int i = 0; i < 8; ++i) r[i] = p[i];
#pragma unroll
  for (int i = 0; i < 8; ++i) r[8 + i] = p[16 + i];
  return r;
}

// B (32x16 KxN) from a row-major KxN source: lane<16 -> col=lane, K 0..7 &
// 16..23; lane>=16 -> col=lane-16, K 8..15 & 24..31.
__device__ __forceinline__ v16h frag_col(const _Float16* base, int ld, int lane) {
  int n  = lane & 15;
  int ko = (lane & 16) ? 8 : 0;
  v16h r;
#pragma unroll
  for (int i = 0; i < 8; ++i) r[i] = base[(ko + i) * ld + n];
#pragma unroll
  for (int i = 0; i < 8; ++i) r[8 + i] = base[(16 + ko + i) * ld + n];
  return r;
}

// block-wide sum; valid in thread 0 only. red = 8-float shared scratch.
__device__ __forceinline__ float block_sum(float v, float* red) {
  int lane = threadIdx.x & 31, w = threadIdx.x >> 5;
#pragma unroll
  for (int m = 16; m; m >>= 1) v += __shfl_xor(v, m, 32);
  if (lane == 0) red[w] = v;
  __syncthreads();
  float r = 0.0f;
  if (threadIdx.x == 0)
    for (int i = 0; i < 8; ++i) r += red[i];
  __syncthreads();
  return r;
}

// ---------------- kernel 0: zero accumulators ----------------
__global__ void zero_acc_kernel(float* ws) {
  int t = threadIdx.x;
  if (t < WS_ACC_N) ws[WS_ACC + t] = 0.0f;
}

// ---------------- kernel 1: row squared norms ----------------
// one wave per row (d=64 -> 2 elems/lane)
__global__ __launch_bounds__(256) void rowsumsq_kernel(const float* __restrict__ Q,
                                                       const float* __restrict__ K,
                                                       float* __restrict__ ws) {
  int lane = threadIdx.x & 31;
  int row  = blockIdx.x * 8 + (threadIdx.x >> 5);
  const float* kp = K + (size_t)row * D_;
  const float* qp = Q + (size_t)row * D_;
  float a = kp[lane], b = kp[lane + 32];
  float ks = a * a + b * b;
  float c = qp[lane], d = qp[lane + 32];
  float qs = c * c + d * d;
#pragma unroll
  for (int m = 16; m; m >>= 1) { ks += __shfl_xor(ks, m, 32); qs += __shfl_xor(qs, m, 32); }
  if (lane == 0) { ws[WS_K2 + row] = ks; ws[WS_Q2 + row] = qs; }
}

// ---------------- kernel 2: dist = sqrt(max(k2+q2-2 K.Q^T,0)) ----------------
// block tile 64(i) x 128(j); 8 waves as 2x4, each wave 32x32 (2x2 WMMA frags).
// Fuses per-head BN1 sum/sumsq accumulation. Writes dist into the cw region.
__global__ __launch_bounds__(256) void dist_kernel(const float* __restrict__ K,
                                                   const float* __restrict__ Q,
                                                   float* __restrict__ ws,
                                                   float* __restrict__ cw) {
  __shared__ _Float16 Kt[64][72];
  __shared__ _Float16 Qt[128][72];
  __shared__ float red[8];

  int bh = blockIdx.z;
  int i0 = blockIdx.y * 64;
  int j0 = blockIdx.x * 128;
  int tid = threadIdx.x;

  const float* Kp = K + ((size_t)bh * L_ + i0) * D_;
  const float* Qp = Q + ((size_t)bh * L_ + j0) * D_;
  for (int e = tid; e < 64 * 64; e += 256)
    Kt[e >> 6][e & 63] = (_Float16)Kp[e];
  for (int e = tid; e < 128 * 64; e += 256)
    Qt[e >> 6][e & 63] = (_Float16)Qp[e];
  __syncthreads();

  int lane = tid & 31, w = tid >> 5;
  int wi = w & 1, wj = w >> 1;

  v8f acc[2][2] = {};
#pragma unroll
  for (int kk = 0; kk < D_; kk += 32) {
    v16h a0 = frag_row(&Kt[wi * 32 + 0][kk], 72, lane);
    v16h a1 = frag_row(&Kt[wi * 32 + 16][kk], 72, lane);
    v16h b0 = frag_row(&Qt[wj * 32 + 0][kk], 72, lane);   // B[k][n] = Q[n][k]
    v16h b1 = frag_row(&Qt[wj * 32 + 16][kk], 72, lane);
    acc[0][0] = WMMA_F16(a0, b0, acc[0][0]);
    acc[0][1] = WMMA_F16(a0, b1, acc[0][1]);
    acc[1][0] = WMMA_F16(a1, b0, acc[1][0]);
    acc[1][1] = WMMA_F16(a1, b1, acc[1][1]);
  }

  const float* k2p = ws + WS_K2 + (size_t)bh * L_;
  const float* q2p = ws + WS_Q2 + (size_t)bh * L_;
  float lsum = 0.0f, lsq = 0.0f;
#pragma unroll
  for (int mi = 0; mi < 2; ++mi)
#pragma unroll
    for (int nj = 0; nj < 2; ++nj) {
      int j = j0 + wj * 32 + nj * 16 + (lane & 15);
      float q2v = q2p[j];
      int ibase = i0 + wi * 32 + mi * 16 + ((lane & 16) ? 8 : 0);
#pragma unroll
      for (int v = 0; v < 8; ++v) {
        int i = ibase + v;
        float d2 = k2p[i] + q2v - 2.0f * acc[mi][nj][v];
        float dist = sqrtf(fmaxf(d2, 0.0f));
        cw[((size_t)bh * L_ + i) * L_ + j] = dist;
        lsum += dist;
        lsq += dist * dist;
      }
    }

  float bs = block_sum(lsum, red);
  if (tid == 0) atomicAdd(&ws[WS_BN1_SUM + bh % H_], bs);
  float bq = block_sum(lsq, red);
  if (tid == 0) atomicAdd(&ws[WS_BN1_SQ + bh % H_], bq);
}

// ---------------- kernel 3: BN1 stat finalize ----------------
__global__ void bn1_finalize_kernel(float* ws) {
  int h = threadIdx.x;
  if (h < H_) {
    float cnt  = (float)((size_t)B_ * L_ * L_);
    float mean = ws[WS_BN1_SUM + h] / cnt;
    float var  = ws[WS_BN1_SQ + h] / cnt - mean * mean;
    ws[WS_BN1_MEAN + h] = mean;
    ws[WS_BN1_RSTD + h] = rsqrtf(fmaxf(var, 0.0f) + 1e-5f);
  }
}

// ---------------- kernel 4: BN affine + leaky(99) + mask + softmax, in place ----
// one block per row (L=1024, 4 elems/thread)
__global__ __launch_bounds__(256) void softmax_kernel(float* __restrict__ cw,
                                                      const unsigned char* __restrict__ pm,
                                                      const float* __restrict__ ws,
                                                      const float* __restrict__ g1,
                                                      const float* __restrict__ b1) {
  __shared__ float red[8];
  __shared__ float bc;

  int row = blockIdx.x;            // bh*L + i
  int h   = (row >> 10) % H_;
  float mean = ws[WS_BN1_MEAN + h];
  float rstd = ws[WS_BN1_RSTD + h];
  float g = g1[h], bb = b1[h];

  float* rp = cw + (size_t)row * L_;
  const unsigned char* mp = pm + (size_t)row * L_;
  int tid = threadIdx.x, lane = tid & 31, w = tid >> 5;

  float x[4];
#pragma unroll
  for (int t = 0; t < 4; ++t) {
    int j = tid + t * 256;
    float y = (rp[j] - mean) * rstd * g + bb;
    y = (y >= 0.0f) ? y : 99.0f * y;          // LeakyReLU(99.0)
    if (mp[j]) y = -__builtin_inff();          // pad mask -> -inf
    x[t] = y;
  }

  float m = fmaxf(fmaxf(x[0], x[1]), fmaxf(x[2], x[3]));
#pragma unroll
  for (int s = 16; s; s >>= 1) m = fmaxf(m, __shfl_xor(m, s, 32));
  if (lane == 0) red[w] = m;
  __syncthreads();
  if (tid == 0) {
    float r = red[0];
    for (int i = 1; i < 8; ++i) r = fmaxf(r, red[i]);
    bc = r;
  }
  __syncthreads();
  float rowmax = bc;

  if (rowmax == -__builtin_inff()) {           // fully masked row: NaN -> 0
#pragma unroll
    for (int t = 0; t < 4; ++t) rp[tid + t * 256] = 0.0f;
    return;
  }

  float e[4], ls = 0.0f;
#pragma unroll
  for (int t = 0; t < 4; ++t) {
    e[t] = (x[t] == -__builtin_inff()) ? 0.0f : expf(x[t] - rowmax);
    ls += e[t];
  }
#pragma unroll
  for (int s = 16; s; s >>= 1) ls += __shfl_xor(ls, s, 32);
  if (lane == 0) red[w] = ls;
  __syncthreads();
  if (tid == 0) {
    float r = 0.0f;
    for (int i = 0; i < 8; ++i) r += red[i];
    bc = r;
  }
  __syncthreads();
  float inv = 1.0f / bc;
#pragma unroll
  for (int t = 0; t < 4; ++t) rp[tid + t * 256] = e[t] * inv;
}

// ---------------- kernel 5: fused O = P x {V,K,Q} ----------------
// block: 128 rows of P x 64 cols; 8 waves, wave w owns rows [16w,16w+16).
// P read once (fp32 -> f16 into LDS) and shared by all three GEMMs.
// Fuses per-head BN2d sum/sumsq accumulation for K_out/Q_out.
__global__ __launch_bounds__(256) void pv_kernel(const float* __restrict__ cw,
                                                 const float* __restrict__ V,
                                                 const float* __restrict__ K,
                                                 const float* __restrict__ Q,
                                                 float* __restrict__ vout,
                                                 float* __restrict__ kout,
                                                 float* __restrict__ qout,
                                                 float* __restrict__ ws) {
  __shared__ _Float16 Pt[128][40];
  __shared__ _Float16 Vt[32][72];
  __shared__ _Float16 Ktt[32][72];
  __shared__ _Float16 Qtt[32][72];
  __shared__ float red[8];

  int bh = blockIdx.y;
  int i0 = blockIdx.x * 128;
  int tid = threadIdx.x, lane = tid & 31, w = tid >> 5;

  const float* Pb = cw + ((size_t)bh * L_ + i0) * L_;
  const float* Vb = V + (size_t)bh * L_ * D_;
  const float* Kb = K + (size_t)bh * L_ * D_;
  const float* Qb = Q + (size_t)bh * L_ * D_;

  v8f accV[4] = {}, accK[4] = {}, accQ[4] = {};

  for (int kk = 0; kk < L_; kk += 32) {
    __syncthreads();
    for (int e = tid; e < 128 * 32; e += 256) {
      int r = e >> 5, c = e & 31;
      Pt[r][c] = (_Float16)Pb[(size_t)r * L_ + kk + c];
    }
    for (int e = tid; e < 32 * 64; e += 256) {
      int r = e >> 6, c = e & 63;
      size_t gi = (size_t)(kk + r) * D_ + c;
      Vt[r][c]  = (_Float16)Vb[gi];
      Ktt[r][c] = (_Float16)Kb[gi];
      Qtt[r][c] = (_Float16)Qb[gi];
    }
    __syncthreads();

    v16h a = frag_row(&Pt[w * 16][0], 40, lane);
#pragma unroll
    for (int nj = 0; nj < 4; ++nj) {
      v16h bv = frag_col(&Vt[0][nj * 16], 72, lane);
      accV[nj] = WMMA_F16(a, bv, accV[nj]);
      v16h bk = frag_col(&Ktt[0][nj * 16], 72, lane);
      accK[nj] = WMMA_F16(a, bk, accK[nj]);
      v16h bq = frag_col(&Qtt[0][nj * 16], 72, lane);
      accQ[nj] = WMMA_F16(a, bq, accQ[nj]);
    }
  }

  float sK = 0.0f, qK = 0.0f, sQ = 0.0f, qQ = 0.0f;
#pragma unroll
  for (int nj = 0; nj < 4; ++nj) {
    int n = nj * 16 + (lane & 15);
    int ibase = i0 + w * 16 + ((lane & 16) ? 8 : 0);
#pragma unroll
    for (int v = 0; v < 8; ++v) {
      size_t o = ((size_t)bh * L_ + ibase + v) * D_ + n;
      vout[o] = accV[nj][v];
      float kv = accK[nj][v];
      kout[o] = kv; sK += kv; qK += kv * kv;
      float qv = accQ[nj][v];
      qout[o] = qv; sQ += qv; qQ += qv * qv;
    }
  }

  int h = bh % H_;
  float t;
  t = block_sum(sK, red); if (tid == 0) atomicAdd(&ws[WS_BNK_SUM + h], t);
  t = block_sum(qK, red); if (tid == 0) atomicAdd(&ws[WS_BNK_SQ + h], t);
  t = block_sum(sQ, red); if (tid == 0) atomicAdd(&ws[WS_BNQ_SUM + h], t);
  t = block_sum(qQ, red); if (tid == 0) atomicAdd(&ws[WS_BNQ_SQ + h], t);
}

// ---------------- kernel 6: BN2d finalize in place (eps = d = 64, kept bug) ----
__global__ __launch_bounds__(256) void bn2_kernel(float* __restrict__ kout,
                                                  float* __restrict__ qout,
                                                  const float* __restrict__ ws,
                                                  const float* __restrict__ gk,
                                                  const float* __restrict__ bk,
                                                  const float* __restrict__ gq,
                                                  const float* __restrict__ bq) {
  size_t idx = (size_t)blockIdx.x * 256 + threadIdx.x;
  if (idx >= (size_t)BH_ * L_ * D_) return;
  int h = (int)((idx >> 16) % H_);            // L_*D_ = 65536
  float cnt = (float)(B_ * L_ * D_);
  float mK = ws[WS_BNK_SUM + h] / cnt;
  float vK = ws[WS_BNK_SQ + h] / cnt - mK * mK;
  float rK = rsqrtf(fmaxf(vK, 0.0f) + (float)D_);
  float mQ = ws[WS_BNQ_SUM + h] / cnt;
  float vQ = ws[WS_BNQ_SQ + h] / cnt - mQ * mQ;
  float rQ = rsqrtf(fmaxf(vQ, 0.0f) + (float)D_);
  kout[idx] = (kout[idx] - mK) * rK * gk[h] + bk[h];
  qout[idx] = (qout[idx] - mQ) * rQ * gq[h] + bq[h];
}

// ---------------- host launch ----------------
extern "C" void kernel_launch(void* const* d_in, const int* in_sizes, int n_in,
                              void* d_out, int out_size, void* d_ws, size_t ws_size,
                              hipStream_t stream) {
  (void)in_sizes; (void)n_in; (void)out_size; (void)ws_size;
  const float* Q = (const float*)d_in[0];
  const float* K = (const float*)d_in[1];
  const float* V = (const float*)d_in[2];
  const unsigned char* pm = (const unsigned char*)d_in[3];
  const float* g1 = (const float*)d_in[4];
  const float* b1 = (const float*)d_in[5];
  const float* gk = (const float*)d_in[6];
  const float* bk = (const float*)d_in[7];
  const float* gq = (const float*)d_in[8];
  const float* bq = (const float*)d_in[9];

  float* out = (float*)d_out;
  float* ws  = (float*)d_ws;

  const size_t cw_sz  = (size_t)BH_ * L_ * L_;
  const size_t om_sz  = (size_t)BH_ * L_ * D_;
  float* cw   = out;
  float* vout = out + cw_sz;
  float* kout = vout + om_sz;
  float* qout = kout + om_sz;

  zero_acc_kernel<<<1, 64, 0, stream>>>(ws);
  rowsumsq_kernel<<<BH_ * L_ / 8, 256, 0, stream>>>(Q, K, ws);
  dist_kernel<<<dim3(L_ / 128, L_ / 64, BH_), 256, 0, stream>>>(K, Q, ws, cw);
  bn1_finalize_kernel<<<1, 32, 0, stream>>>(ws);
  softmax_kernel<<<BH_ * L_, 256, 0, stream>>>(cw, pm, ws, g1, b1);
  pv_kernel<<<dim3(L_ / 128, BH_), 256, 0, stream>>>(cw, V, K, Q, vout, kout, qout, ws);
  bn2_kernel<<<(int)((om_sz + 255) / 256), 256, 0, stream>>>(kout, qout, ws, gk, bk, gq, bq);
}
